// DecoderRNN2D_1949915153061
// MI455X (gfx1250) — compile-verified, matrix operand order checked by reading, not run on hardware
//
#include <hip/hip_runtime.h>
#include <hip/hip_bf16.h>
#include <cstdint>
#include <cstddef>

#define DEV __device__ __forceinline__

typedef __bf16 bhalf;
typedef __attribute__((ext_vector_type(16))) __bf16 v16bf;
typedef __attribute__((ext_vector_type(8)))  float    v8f;

namespace {
constexpr int kB    = 256;          // batch
constexpr int kH    = 512;          // hidden
constexpr int kEMB  = 256;
constexpr int kCF   = 512;
constexpr int kHW   = 64;           // 8x8
constexpr int kPAD  = 100;          // 10x10 zero-padded spatial (halo for 3x3 conv)
constexpr int kNMIX = 20;
constexpr int kTS   = 257;          // scan steps (start token + 256)
constexpr int kKG   = 1056;         // gates GEMM K: 517 (att|pt) + 512 (h) padded to 33*32
constexpr int kNG   = 2048;         // gates GEMM N (4*H)
constexpr int kNP   = 128;          // output projection N: 123 padded to 128
constexpr int kMO   = kB * kTS;     // 65792 output rows
}

// ---------- bf16 helpers (round-to-nearest-even) ----------
DEV bhalf f2bf(float x) {
  union { float f; uint32_t u; } v; v.f = x;
  uint32_t r = v.u + 0x7FFFu + ((v.u >> 16) & 1u);
  uint16_t h = (uint16_t)(r >> 16);
  bhalf out; __builtin_memcpy(&out, &h, 2); return out;
}

// ---------- WMMA fragment helpers (CDNA5 16x16x32 bf16 layout) ----------
// 16-bit A (16x32): lanes 0-15 rows M, lanes 16-31 same rows with K+8;
// per lane: elems 0..7 -> K..K+7, elems 8..15 -> K+16..K+23.
// B (32x16) is symmetric with lane -> column N.
DEV v16bf frag_load(const bhalf* __restrict__ row, int k) {
  const bhalf* p = row + k;
  v16bf f;
#pragma unroll
  for (int j = 0; j < 8; ++j) { f[j] = p[j]; f[8 + j] = p[16 + j]; }
  return f;
}

DEV v8f acc_zero() {
  v8f a;
#pragma unroll
  for (int j = 0; j < 8; ++j) a[j] = 0.f;
  return a;
}

DEV v8f wmma_bf16(v16bf a, v16bf b, v8f c) {
  // (neg_a, A, neg_b, B, c_mod, C, reuse_a, reuse_b)
  return __builtin_amdgcn_wmma_f32_16x16x32_bf16(false, a, false, b, (short)0, c, false, false);
}

// ---------- prep kernels (per-launch weight conversion / rearrangement) ----------
// backbone (B,C,8,8) f32 -> zero-padded (B,10,10,C) bf16 halo buffer
__global__ void k_prep_orgpad(const float* __restrict__ src, bhalf* __restrict__ dst) {
  int idx = blockIdx.x * blockDim.x + threadIdx.x;   // b*100*CF + pos*CF + c
  if (idx >= kB * kPAD * kCF) return;
  int c = idx & 511;
  int t = idx >> 9;
  int pos = t % kPAD, b = t / kPAD;
  int y = pos / 10, x = pos % 10;
  int ys = y - 1, xs = x - 1;
  float v = 0.f;
  if ((unsigned)ys < 8u && (unsigned)xs < 8u)
    v = src[((size_t)b * kCF + c) * kHW + ys * 8 + xs];
  dst[idx] = f2bf(v);
}

__global__ void k_prep_kw(const float* __restrict__ src, bhalf* __restrict__ dst) {
  int idx = blockIdx.x * blockDim.x + threadIdx.x;   // s*EMB*CF + e*CF + c
  int c = idx & 511, e = (idx >> 9) & 255, s = idx >> 17;
  dst[idx] = f2bf(src[((size_t)e * kCF + c) * 9 + s]);
}

__global__ void k_prep_bf(const float* __restrict__ src, bhalf* __restrict__ dst, int n) {
  int idx = blockIdx.x * blockDim.x + threadIdx.x;
  if (idx < n) dst[idx] = f2bf(src[idx]);
}

__global__ void k_prep_wg(const float* __restrict__ wih, const float* __restrict__ whh,
                          bhalf* __restrict__ dst) {
  int idx = blockIdx.x * blockDim.x + threadIdx.x;   // n*kKG + k
  int k = idx % kKG, n = idx / kKG;
  float v;
  if (k < 517)       v = wih[(size_t)n * 517 + k];
  else if (k < 1029) v = whh[(size_t)n * 512 + (k - 517)];
  else               v = 0.f;
  dst[idx] = f2bf(v);
}

__global__ void k_prep_wp(const float* __restrict__ src, bhalf* __restrict__ dst) {
  int idx = blockIdx.x * blockDim.x + threadIdx.x;   // n*kH + k, n < 128
  int k = idx & 511, n = idx >> 9;
  dst[idx] = f2bf((n < 123) ? src[(size_t)n * kH + k] : 0.f);
}

__global__ void k_abuf_pad(bhalf* __restrict__ abuf) {
  int idx = blockIdx.x * blockDim.x + threadIdx.x;
  if (idx >= kB * (kKG - 1029)) return;
  int b = idx / 27, k = 1029 + idx % 27;
  abuf[(size_t)b * kKG + k] = f2bf(0.f);
}

// ---------- h0/c0 = tanh(z @ W_fc_hc^T + b) ----------
__global__ void k_init_hc(const float* __restrict__ z, const float* __restrict__ w,
                          const float* __restrict__ bias, bhalf* __restrict__ hbf,
                          float* __restrict__ cst) {
  int idx = blockIdx.x * blockDim.x + threadIdx.x;   // < kB * 2H
  int b = idx >> 10, o = idx & 1023;
  const float* zr = z + (size_t)b * 128;
  const float* wr = w + (size_t)o * 128;
  float s = bias[o];
#pragma unroll 8
  for (int k = 0; k < 128; ++k) s += zr[k] * wr[k];
  float v = tanhf(s);
  if (o < kH) hbf[(size_t)b * kH + o] = f2bf(v);
  else        cst[(size_t)b * kH + (o - kH)] = v;
}

// ---------- conv as 9 shifted WMMA GEMMs over the halo buffer ----------
// 1x4 register blocking: 1 A fragment feeds 4 independent accumulators.
__global__ void k_conv(const bhalf* __restrict__ orgp, const bhalf* __restrict__ kw,
                       const float* __restrict__ bcf, float* __restrict__ xem) {
  int wave = (blockIdx.x * blockDim.x + threadIdx.x) >> 5;
  int lane = threadIdx.x & 31;
  int mtile = wave >> 2, ngrp = wave & 3;            // 1024 x 4
  int row = mtile * 16 + (lane & 15);
  int bb = row >> 6, p = row & 63, y = p >> 3, x = p & 7;
  int kb = (lane >> 4) << 3;
  const bhalf* brow0 = kw + (size_t)(ngrp * 64 + (lane & 15)) * kCF;
  v8f acc[4];
#pragma unroll
  for (int i = 0; i < 4; ++i) acc[i] = acc_zero();
  for (int s = 0; s < 9; ++s) {
    // halo coords: padded (y + dy, x + dx) with dy,dx in 0..2
    const bhalf* arow = orgp + ((size_t)bb * kPAD + (y + s / 3) * 10 + (x + s % 3)) * kCF;
    const bhalf* brow = brow0 + (size_t)s * kEMB * kCF;
#pragma unroll 2
    for (int k0 = 0; k0 < kCF; k0 += 32) {
      v16bf a = frag_load(arow, k0 + kb);
#pragma unroll
      for (int i = 0; i < 4; ++i)
        acc[i] = wmma_bf16(a, frag_load(brow + (size_t)i * 16 * kCF, k0 + kb), acc[i]);
    }
  }
  int m0 = mtile * 16 + ((lane >> 4) << 3);
#pragma unroll
  for (int i = 0; i < 4; ++i) {
    int n = ngrp * 64 + i * 16 + (lane & 15);
    float bi = bcf[n];
#pragma unroll
    for (int v = 0; v < 8; ++v) xem[(size_t)(m0 + v) * kEMB + n] = acc[i][v] + bi;
  }
}

// ---------- g_em = h @ W_conv_h^T + b (WMMA, 256x256, 1x4 blocking) ----------
__global__ void k_gem(const bhalf* __restrict__ hbf, const bhalf* __restrict__ wch,
                      const float* __restrict__ bch, float* __restrict__ gem) {
  int wave = (blockIdx.x * blockDim.x + threadIdx.x) >> 5;
  int lane = threadIdx.x & 31;
  int mtile = wave >> 2, ngrp = wave & 3;            // 16 x 4
  const bhalf* arow = hbf + (size_t)(mtile * 16 + (lane & 15)) * kH;
  const bhalf* brow = wch + (size_t)(ngrp * 64 + (lane & 15)) * kH;
  int kb = (lane >> 4) << 3;
  v8f acc[4];
#pragma unroll
  for (int i = 0; i < 4; ++i) acc[i] = acc_zero();
#pragma unroll 2
  for (int k0 = 0; k0 < kH; k0 += 32) {
    v16bf a = frag_load(arow, k0 + kb);
#pragma unroll
    for (int i = 0; i < 4; ++i)
      acc[i] = wmma_bf16(a, frag_load(brow + (size_t)i * 16 * kH, k0 + kb), acc[i]);
  }
  int m0 = mtile * 16 + ((lane >> 4) << 3);
#pragma unroll
  for (int i = 0; i < 4; ++i) {
    int n = ngrp * 64 + i * 16 + (lane & 15);
    float bi = bch[n];
#pragma unroll
    for (int v = 0; v < 8; ++v) gem[(size_t)(m0 + v) * kEMB + n] = acc[i][v] + bi;
  }
}

// ---------- attention: scores, softmax, weighted sum; builds fused A row ----------
__global__ void k_att(const float* __restrict__ xem, const float* __restrict__ gem,
                      const float* __restrict__ watt, const float* __restrict__ batt,
                      const float* __restrict__ backbone, const float* __restrict__ sketch,
                      const bhalf* __restrict__ hbf, bhalf* __restrict__ abuf, int t) {
  __shared__ float s_sc[kHW];
  int b = blockIdx.x;
  int tid = threadIdx.x, lane = tid & 31, wid = tid >> 5;
  const float* xe = xem + (size_t)b * kHW * kEMB;
  const float* ge = gem + (size_t)b * kEMB;
  for (int p = wid; p < kHW; p += 8) {
    const float* xr = xe + (size_t)p * kEMB;
    float s = 0.f;
    for (int e = lane; e < kEMB; e += 32) s += tanhf(xr[e] + ge[e]) * watt[e];
#pragma unroll
    for (int off = 16; off > 0; off >>= 1) s += __shfl_down(s, off, 32);
    if (lane == 0) s_sc[p] = s + batt[0];
  }
  __syncthreads();
  if (wid == 0) {
    float v0 = s_sc[lane], v1 = s_sc[lane + 32];
    float m = fmaxf(v0, v1);
#pragma unroll
    for (int off = 16; off > 0; off >>= 1) m = fmaxf(m, __shfl_xor(m, off, 32));
    float e0 = expf(v0 - m), e1 = expf(v1 - m);
    float s = e0 + e1;
#pragma unroll
    for (int off = 16; off > 0; off >>= 1) s += __shfl_xor(s, off, 32);
    float inv = 1.f / s;
    s_sc[lane] = e0 * inv; s_sc[lane + 32] = e1 * inv;
  }
  __syncthreads();
  bhalf* arow = abuf + (size_t)b * kKG;
  for (int c = tid; c < kCF; c += 256) {
    const float* src = backbone + ((size_t)b * kCF + c) * kHW;
    float a = 0.f;
#pragma unroll 8
    for (int p = 0; p < kHW; ++p) a += s_sc[p] * src[p];
    arow[c] = f2bf(a);
  }
  if (tid < 5) {
    float pv = (t == 0) ? ((tid == 2) ? 1.f : 0.f)
                        : sketch[((size_t)(t - 1) * kB + b) * 5 + tid];
    arow[kCF + tid] = f2bf(pv);
  }
  for (int k2 = tid; k2 < kH; k2 += 256) arow[517 + k2] = hbf[(size_t)b * kH + k2];
}

// ---------- gates = [att|pt|h] @ Wg^T (WMMA, 256x2048, K=1056, 1x4 blocking) ----------
__global__ void k_gates(const bhalf* __restrict__ abuf, const bhalf* __restrict__ wg,
                        float* __restrict__ gates) {
  int wave = (blockIdx.x * blockDim.x + threadIdx.x) >> 5;
  int lane = threadIdx.x & 31;
  int mtile = wave >> 5, ngrp = wave & 31;           // 16 x 32
  const bhalf* arow = abuf + (size_t)(mtile * 16 + (lane & 15)) * kKG;
  const bhalf* brow = wg + (size_t)(ngrp * 64 + (lane & 15)) * kKG;
  int kb = (lane >> 4) << 3;
  v8f acc[4];
#pragma unroll
  for (int i = 0; i < 4; ++i) acc[i] = acc_zero();
  for (int k0 = 0; k0 < kKG; k0 += 32) {
    __builtin_prefetch(brow + k0 + 256, 0, 1);       // global_prefetch_b8 on weight stream
    v16bf a = frag_load(arow, k0 + kb);
#pragma unroll
    for (int i = 0; i < 4; ++i)
      acc[i] = wmma_bf16(a, frag_load(brow + (size_t)i * 16 * kKG, k0 + kb), acc[i]);
  }
  int m0 = mtile * 16 + ((lane >> 4) << 3);
#pragma unroll
  for (int i = 0; i < 4; ++i) {
    int n = ngrp * 64 + i * 16 + (lane & 15);
#pragma unroll
    for (int v = 0; v < 8; ++v) gates[(size_t)(m0 + v) * kNG + n] = acc[i][v];
  }
}

// ---------- LSTM pointwise: c/h update, append hidden ----------
__global__ void k_lstm(const float* __restrict__ gates, const float* __restrict__ bih,
                       const float* __restrict__ bhh, float* __restrict__ cst,
                       bhalf* __restrict__ hbf, bhalf* __restrict__ hid_t) {
  int idx = blockIdx.x * blockDim.x + threadIdx.x;   // < kB*kH
  int b = idx >> 9, j = idx & 511;
  const float* g = gates + (size_t)b * kNG;
  float gi = g[j]        + bih[j]        + bhh[j];
  float gf = g[512 + j]  + bih[512 + j]  + bhh[512 + j];
  float gg = g[1024 + j] + bih[1024 + j] + bhh[1024 + j];
  float go = g[1536 + j] + bih[1536 + j] + bhh[1536 + j];
  float si = 1.f / (1.f + expf(-gi));
  float sf = 1.f / (1.f + expf(-gf));
  float so = 1.f / (1.f + expf(-go));
  float cn = sf * cst[idx] + si * tanhf(gg);
  float hn = so * tanhf(cn);
  cst[idx] = cn;
  bhalf hb = f2bf(hn);
  hbf[idx]   = hb;
  hid_t[idx] = hb;
}

// ---------- y = hiddens @ W_fc_params^T (WMMA, 65792 x 128, K=512, 1x4) ----------
__global__ void k_out(const bhalf* __restrict__ hid, const bhalf* __restrict__ wp,
                      const float* __restrict__ bp, float* __restrict__ y) {
  int wave = (blockIdx.x * blockDim.x + threadIdx.x) >> 5;
  int lane = threadIdx.x & 31;
  int mtile = wave >> 1, ngrp = wave & 1;            // 4112 x 2
  int row = mtile * 16 + (lane & 15);                // row = b*257 + t
  int b = row / kTS, t = row % kTS;
  const bhalf* arow = hid + ((size_t)t * kB + b) * kH;
  const bhalf* brow = wp + (size_t)(ngrp * 64 + (lane & 15)) * kH;
  int kb = (lane >> 4) << 3;
  v8f acc[4];
#pragma unroll
  for (int i = 0; i < 4; ++i) acc[i] = acc_zero();
#pragma unroll 2
  for (int k0 = 0; k0 < kH; k0 += 32) {
    v16bf a = frag_load(arow, k0 + kb);
#pragma unroll
    for (int i = 0; i < 4; ++i)
      acc[i] = wmma_bf16(a, frag_load(brow + (size_t)i * 16 * kH, k0 + kb), acc[i]);
  }
  int m0 = mtile * 16 + ((lane >> 4) << 3);
#pragma unroll
  for (int i = 0; i < 4; ++i) {
    int n = ngrp * 64 + i * 16 + (lane & 15);
    float bi = (n < 123) ? bp[n] : 0.f;
#pragma unroll
    for (int v = 0; v < 8; ++v) y[(size_t)(m0 + v) * kNP + n] = acc[i][v] + bi;
  }
}

// ---------- MDN head: softmax(pi), exp(s), tanh(corr), pen logits ----------
__global__ void k_finalize(const float* __restrict__ y, float* __restrict__ out) {
  int r = blockIdx.x * blockDim.x + threadIdx.x;
  if (r >= kMO) return;
  const float* yr = y + (size_t)r * kNP;
  const size_t S = (size_t)kMO * kNMIX;
  float* pi   = out;
  float* mu1  = out + S;
  float* mu2  = out + 2 * S;
  float* s1   = out + 3 * S;
  float* s2   = out + 4 * S;
  float* corr = out + 5 * S;
  float* pen  = out + 6 * S;
  pen[(size_t)r * 3 + 0] = yr[0];
  pen[(size_t)r * 3 + 1] = yr[1];
  pen[(size_t)r * 3 + 2] = yr[2];
  float e[kNMIX];
  float m = -3.402823466e38f;
#pragma unroll
  for (int i = 0; i < kNMIX; ++i) m = fmaxf(m, yr[3 + i]);
  float s = 0.f;
#pragma unroll
  for (int i = 0; i < kNMIX; ++i) { e[i] = expf(yr[3 + i] - m); s += e[i]; }
  float inv = 1.f / s;
#pragma unroll
  for (int i = 0; i < kNMIX; ++i) {
    pi[(size_t)r * kNMIX + i]   = e[i] * inv;
    mu1[(size_t)r * kNMIX + i]  = yr[23 + i];
    mu2[(size_t)r * kNMIX + i]  = yr[43 + i];
    s1[(size_t)r * kNMIX + i]   = expf(yr[63 + i]);
    s2[(size_t)r * kNMIX + i]   = expf(yr[83 + i]);
    corr[(size_t)r * kNMIX + i] = tanhf(yr[103 + i]);
  }
}

extern "C" void kernel_launch(void* const* d_in, const int* in_sizes, int n_in,
                              void* d_out, int out_size, void* d_ws, size_t ws_size,
                              hipStream_t stream) {
  (void)in_sizes; (void)n_in; (void)out_size; (void)ws_size;
  const float* backbone = (const float*)d_in[0];
  const float* zvec     = (const float*)d_in[1];
  const float* sketch   = (const float*)d_in[2];
  const float* Wfc      = (const float*)d_in[3];
  const float* bfc      = (const float*)d_in[4];
  const float* Wch      = (const float*)d_in[5];
  const float* bch      = (const float*)d_in[6];
  const float* Kcf      = (const float*)d_in[7];
  const float* bcf      = (const float*)d_in[8];
  const float* Watt     = (const float*)d_in[9];
  const float* batt     = (const float*)d_in[10];
  const float* Wih      = (const float*)d_in[11];
  const float* Whh      = (const float*)d_in[12];
  const float* bih      = (const float*)d_in[13];
  const float* bhh      = (const float*)d_in[14];
  const float* Wp       = (const float*)d_in[15];
  const float* bp       = (const float*)d_in[16];
  float* out = (float*)d_out;

  char* cur = (char*)d_ws;
  auto take = [&](size_t bytes) -> char* {
    char* p = cur; cur += (bytes + 255) & ~(size_t)255; return p;
  };
  bhalf* orgp   = (bhalf*)take((size_t)kB * kPAD * kCF * 2);  // zero-padded halo (B,10,10,C)
  bhalf* kw     = (bhalf*)take((size_t)9 * kEMB * kCF * 2);   // per-tap conv weights
  bhalf* wchb   = (bhalf*)take((size_t)kEMB * kH * 2);
  bhalf* wg     = (bhalf*)take((size_t)kNG * kKG * 2);        // fused [W_ih | W_hh | pad]
  bhalf* wpb    = (bhalf*)take((size_t)kNP * kH * 2);
  float* xem    = (float*)take((size_t)kB * kHW * kEMB * 4);
  float* gem    = (float*)take((size_t)kB * kEMB * 4);
  bhalf* abuf   = (bhalf*)take((size_t)kB * kKG * 2);         // fused GEMM A rows
  float* gates  = (float*)take((size_t)kB * kNG * 4);
  float* cstate = (float*)take((size_t)kB * kH * 4);
  bhalf* hbf    = (bhalf*)take((size_t)kB * kH * 2);
  bhalf* hid    = (bhalf*)take((size_t)kTS * kB * kH * 2);    // all hiddens (t,b,h)
  float* ybuf   = (float*)take((size_t)kMO * kNP * 4);

  dim3 blk(256);
  k_prep_orgpad<<<(kB * kPAD * kCF + 255) / 256, blk, 0, stream>>>(backbone, orgp);
  k_prep_kw<<<(9 * kEMB * kCF) / 256, blk, 0, stream>>>(Kcf, kw);
  k_prep_bf<<<(kEMB * kH) / 256, blk, 0, stream>>>(Wch, wchb, kEMB * kH);
  k_prep_wg<<<(kNG * kKG) / 256, blk, 0, stream>>>(Wih, Whh, wg);
  k_prep_wp<<<(kNP * kH) / 256, blk, 0, stream>>>(Wp, wpb);
  k_abuf_pad<<<(kB * (kKG - 1029) + 255) / 256, blk, 0, stream>>>(abuf);
  k_init_hc<<<(kB * 2 * kH) / 256, blk, 0, stream>>>(zvec, Wfc, bfc, hbf, cstate);
  k_conv<<<((kB * kHW / 16) * (kEMB / 64)) / 8, blk, 0, stream>>>(orgp, kw, bcf, xem);

  for (int t = 0; t < kTS; ++t) {
    k_gem<<<((kB / 16) * (kEMB / 64)) / 8, blk, 0, stream>>>(hbf, wchb, bch, gem);
    k_att<<<kB, blk, 0, stream>>>(xem, gem, Watt, batt, backbone, sketch, hbf, abuf, t);
    k_gates<<<((kB / 16) * (kNG / 64)) / 8, blk, 0, stream>>>(abuf, wg, gates);
    k_lstm<<<(kB * kH) / 256, blk, 0, stream>>>(gates, bih, bhh, cstate, hbf,
                                                hid + (size_t)t * kB * kH);
  }
  k_out<<<((kMO / 16) * (kNP / 64)) / 8, blk, 0, stream>>>(hid, wpb, bp, ybuf);
  k_finalize<<<(kMO + 255) / 256, blk, 0, stream>>>(ybuf, out);
}
